// LSTM_34969623724677
// MI455X (gfx1250) — matrix operand order repaired and checked by reading, not compiled, not run on hardware
//
#include <hip/hip_runtime.h>
#include <math.h>

#define T_SEQ  2048
#define HID    2048
#define G4H    8192   // 4*HID
#define VOCAB  128
#define HN_OFF (T_SEQ * VOCAB)          // 262144: start of h_n in d_out
#define CN_OFF (HN_OFF + 2 * HID)       // start of c_n in d_out

typedef __attribute__((ext_vector_type(16))) __bf16        v16bf;
typedef __attribute__((ext_vector_type(8)))  float         v8f;
typedef __attribute__((ext_vector_type(4)))  unsigned int  uint4v;
typedef int i4v __attribute__((vector_size(16)));   // matches builtin prototype

union ABfrag { uint4v q[2]; v16bf bf; };

#define AS1 __attribute__((address_space(1)))
#define AS3 __attribute__((address_space(3)))

#if defined(__has_builtin)
#if __has_builtin(__builtin_amdgcn_global_load_async_to_lds_b128)
#define HAVE_ASYNC_LDS 1
#endif
#endif
#ifndef HAVE_ASYNC_LDS
#define HAVE_ASYNC_LDS 0
#endif

// ---------- helpers ----------
__device__ __forceinline__ unsigned short f2bf(float f) {
  union { float f; unsigned int u; } v; v.f = f;
  unsigned int u = v.u;
  u += 0x7fffu + ((u >> 16) & 1u);     // round-to-nearest-even
  return (unsigned short)(u >> 16);
}
__device__ __forceinline__ float bflo(unsigned int u) {
  union { unsigned int u; float f; } v; v.u = u << 16; return v.f;
}
__device__ __forceinline__ float bfhi(unsigned int u) {
  union { unsigned int u; float f; } v; v.u = u & 0xffff0000u; return v.f;
}
__device__ __forceinline__ float sigmoidf_(float x) { return 1.0f / (1.0f + expf(-x)); }

__device__ __forceinline__ void wait_async0() {
#if HAVE_ASYNC_LDS
#if __has_builtin(__builtin_amdgcn_s_wait_asynccnt)
  __builtin_amdgcn_s_wait_asynccnt(0);
#else
  asm volatile("s_wait_asynccnt 0x0" ::: "memory");
#endif
#endif
}

// Device-wide sense-reversal barrier: bar[0]=arrive counter, bar[1]=generation.
__device__ __forceinline__ void grid_sync(unsigned int* bar, unsigned int nblocks) {
  __threadfence();
  __syncthreads();
  if (threadIdx.x == 0) {
    unsigned int gen = __atomic_load_n(bar + 1, __ATOMIC_ACQUIRE);
    if (atomicAdd(bar, 1u) == nblocks - 1u) {
      __atomic_store_n(bar, 0u, __ATOMIC_RELAXED);
      __atomic_fetch_add(bar + 1, 1u, __ATOMIC_RELEASE);
    } else {
      while (__atomic_load_n(bar + 1, __ATOMIC_ACQUIRE) == gen)
        __builtin_amdgcn_s_sleep(1);
    }
  }
  __syncthreads();
}

// ---------- fp32 -> bf16 ----------
__global__ void f32_to_bf16_k(const float* __restrict__ in,
                              unsigned short* __restrict__ out, long long n) {
  long long i = (long long)blockIdx.x * blockDim.x + threadIdx.x;
  long long stride = (long long)gridDim.x * blockDim.x;
  for (; i < n; i += stride) out[i] = f2bf(in[i]);
}

// ---------- WMMA GEMM: C[M,N] = A[M,K] * B[N,K]^T + bias0 + bias1 ----------
// A,B row-major bf16. Each wave computes a 32x64 tile; all waves of a block
// share the same 32 A-rows, which are staged through LDS in 128-wide K chunks,
// double-buffered via async-to-LDS DMA so the copy of chunk k+1 overlaps the
// WMMAs of chunk k. Requires (blockDim/32) to divide N/64.
// __launch_bounds__(256,1): favor VGPRs over occupancy so the pipelined
// B fragments don't spill to scratch.
#define KC 128
__global__ void __launch_bounds__(256, 1)
wmma_gemm_bias(const unsigned short* __restrict__ A,
               const unsigned short* __restrict__ B,
               const float* __restrict__ bias0,
               const float* __restrict__ bias1,
               float* __restrict__ C, int M, int N, int K) {
  __shared__ __attribute__((aligned(16))) unsigned short asmem[2][32][136];

  const int tid  = threadIdx.x;
  const int lane = tid & 31;
  const int wave = tid >> 5;
  const int wpb  = blockDim.x >> 5;                  // waves per block
  const int wn   = N >> 6;                           // 64-col wave tiles
  const int gw0  = blockIdx.x * wpb;
  const int m0   = (gw0 / wn) * 32;                  // shared across block
  const int n0   = ((gw0 + wave) % wn) * 64;

  // A frag layout (16x32 bf16): lanes 0-15 row M=lane, K chunks {0..7,16..23};
  // lanes 16-31 same rows, K chunks {8..15,24..31}.
  const int akb = (lane >> 4) << 3;                  // 0 or 8
  // B frag layout (32x16): lane holds column N=lane&15; lanes 0-15 K {0..7,16..23},
  // lanes 16-31 K {8..15,24..31} (same as A).
  const int bkb = (lane >> 4) << 3;                  // 0 or 8
  const unsigned short* bp = B + (size_t)(n0 + (lane & 15)) * K + bkb;

  // Stage one 32 x KC chunk of A into LDS buffer `buf` (512 x 16B segments).
  auto stage = [&](int kc, int buf) {
    for (int idx = tid; idx < 512; idx += blockDim.x) {
      const int row = idx >> 4;
      const int seg = idx & 15;
      const unsigned short* gp = A + (size_t)(m0 + row) * K + kc + seg * 8;
      unsigned short* lp = &asmem[buf][row][seg * 8];
#if HAVE_ASYNC_LDS
      // AS1: flat/global share representation. AS3: flat LDS addr -> LDS
      // offset is the low 32 bits (ISA 10.2 aperture mapping).
      __builtin_amdgcn_global_load_async_to_lds_b128(
          (AS1 i4v*)(unsigned long long)gp,
          (AS3 i4v*)(unsigned int)(unsigned long long)lp, 0, 0);
#else
      *(uint4v*)lp = *(const uint4v*)gp;
#endif
    }
  };

  const int nkc = K / KC;
  stage(0, 0);
  wait_async0();
  __syncthreads();

  v8f acc[2][4] = {};
  for (int kc = 0; kc < nkc; ++kc) {
    const int cur = kc & 1;
    if (kc + 1 < nkc) stage((kc + 1) * KC, cur ^ 1);  // overlap DMA with WMMA
    const int kcb = kc * KC;
#pragma unroll
    for (int ks = 0; ks < KC; ks += 32) {
      // ---- load phase: A frags from LDS, all 4 B frags from global ----
      ABfrag a0, a1, b[4];
      const unsigned short* l0 = &asmem[cur][lane & 15][ks + akb];
      const unsigned short* l1 = &asmem[cur][16 + (lane & 15)][ks + akb];
      a0.q[0] = *(const uint4v*)(l0);
      a0.q[1] = *(const uint4v*)(l0 + 16);
      a1.q[0] = *(const uint4v*)(l1);
      a1.q[1] = *(const uint4v*)(l1 + 16);
#pragma unroll
      for (int j = 0; j < 4; ++j) {
        const unsigned short* bj = bp + (size_t)(j * 16) * K + kcb + ks;
        b[j].q[0] = *(const uint4v*)(bj);
        b[j].q[1] = *(const uint4v*)(bj + 16);
      }
      // ---- compute phase: 8 back-to-back WMMAs ----
#pragma unroll
      for (int j = 0; j < 4; ++j) {
        acc[0][j] = __builtin_amdgcn_wmma_f32_16x16x32_bf16(
            false, a0.bf, false, b[j].bf, (short)0, acc[0][j], false, false);
        acc[1][j] = __builtin_amdgcn_wmma_f32_16x16x32_bf16(
            false, a1.bf, false, b[j].bf, (short)0, acc[1][j], false, false);
      }
    }
    wait_async0();
    __syncthreads();
  }

  // C/D layout: lane -> N=lane&15; VGPR r -> M = r + 8*(lane>=16)
  const int mrow = (lane >> 4) << 3;
  const int ncol = lane & 15;
#pragma unroll
  for (int s = 0; s < 2; ++s) {
#pragma unroll
    for (int j = 0; j < 4; ++j) {
      const int n = n0 + j * 16 + ncol;
      float bs = (bias0 ? bias0[n] : 0.0f) + (bias1 ? bias1[n] : 0.0f);
      float* cp = C + (size_t)(m0 + s * 16 + mrow) * N + n;
#pragma unroll
      for (int r = 0; r < 8; ++r) cp[(size_t)r * N] = acc[s][j][r] + bs;
    }
  }
}

// ---------- persistent LSTM scan ----------
// 256 blocks x 256 threads; block owns 8 hidden units (1 unit per wave32).
// Block's W_hh slice (8 units x 4 gates x 2048 bf16 = 128KB) is preloaded into
// LDS once; per step only the 4KB hidden vector is re-read from global.
#define SCAN_BLOCKS 256
#define SCAN_LDS    (131072 + 4096)
__global__ void lstm_scan(const float* __restrict__ G,            // [T,8192]
                          const unsigned short* __restrict__ Whh, // [8192,2048] bf16
                          const float* __restrict__ h0,
                          const float* __restrict__ c0,
                          unsigned short* __restrict__ hbf,       // [2048]
                          unsigned short* __restrict__ seq,       // [T,2048]
                          float* __restrict__ hn_out,
                          float* __restrict__ cn_out,
                          unsigned int* __restrict__ bar,
                          int Tlen) {
  extern __shared__ unsigned char smem[];
  unsigned short* wl  = (unsigned short*)smem;            // 64K halves (128KB)
  unsigned short* hsh = (unsigned short*)(smem + 131072); // 2048 halves (4KB)

  const int tid  = threadIdx.x;
  const int lane = tid & 31;
  const int uw   = tid >> 5;                 // wave id == local unit
  const int unit = blockIdx.x * 8 + uw;

  for (int idx = tid; idx < 8192; idx += 256) {
    const int row_local = idx >> 8;          // 0..31 : localUnit*4 + gate
    const int chunk     = idx & 255;
    const int u = row_local >> 2;
    const int g = row_local & 3;
    const long long grow = (long long)g * HID + blockIdx.x * 8 + u;
    *(uint4v*)&wl[row_local * HID + chunk * 8] =
        *(const uint4v*)&Whh[grow * HID + chunk * 8];
  }
  if (blockIdx.x == 0)
    for (int i = tid; i < HID; i += 256) hbf[i] = f2bf(h0[i]);
  float c_val = (lane == 0) ? c0[unit] : 0.0f;
  grid_sync(bar, gridDim.x);

  for (int t = 0; t < Tlen; ++t) {
    *(uint4v*)&hsh[tid * 8] = *(const uint4v*)&hbf[tid * 8];
    grid_sync(bar, gridDim.x);               // everyone snapshotted before writes

    float a0 = 0.f, a1 = 0.f, a2 = 0.f, a3 = 0.f;
    const unsigned short* w0 = &wl[(uw * 4 + 0) * HID];
    const unsigned short* w1 = &wl[(uw * 4 + 1) * HID];
    const unsigned short* w2 = &wl[(uw * 4 + 2) * HID];
    const unsigned short* w3 = &wl[(uw * 4 + 3) * HID];
#pragma unroll 4
    for (int i = 0; i < 32; ++i) {
      const int kp = (lane + (i << 5)) << 1;           // bank-conflict-free dword
      const unsigned int hq = *(const unsigned int*)&hsh[kp];
      const float hA = bflo(hq), hB = bfhi(hq);
      unsigned int w;
      w = *(const unsigned int*)&w0[kp]; a0 += hA * bflo(w) + hB * bfhi(w);
      w = *(const unsigned int*)&w1[kp]; a1 += hA * bflo(w) + hB * bfhi(w);
      w = *(const unsigned int*)&w2[kp]; a2 += hA * bflo(w) + hB * bfhi(w);
      w = *(const unsigned int*)&w3[kp]; a3 += hA * bflo(w) + hB * bfhi(w);
    }
#pragma unroll
    for (int off = 16; off > 0; off >>= 1) {
      a0 += __shfl_down(a0, off, 32);
      a1 += __shfl_down(a1, off, 32);
      a2 += __shfl_down(a2, off, 32);
      a3 += __shfl_down(a3, off, 32);
    }
    if (lane == 0) {
      const float* gr = G + (long long)t * G4H;
      const float ig = sigmoidf_(gr[unit] + a0);
      const float fg = sigmoidf_(gr[HID + unit] + a1);
      const float gg = tanhf(gr[2 * HID + unit] + a2);
      const float og = sigmoidf_(gr[3 * HID + unit] + a3);
      c_val = fg * c_val + ig * gg;
      const float hv = og * tanhf(c_val);
      const unsigned short hb = f2bf(hv);
      hbf[unit] = hb;
      seq[(long long)t * HID + unit] = hb;
      if (t == Tlen - 1) { hn_out[unit] = hv; cn_out[unit] = c_val; }
    }
    grid_sync(bar, gridDim.x);               // publish new h
  }
}

// ---------- log-softmax over V=128, one block per row ----------
__global__ void logsoftmax128(const float* __restrict__ logits,
                              float* __restrict__ out) {
  __shared__ float sm[VOCAB];
  const int t = blockIdx.x, v = threadIdx.x;
  const float x = logits[(size_t)t * VOCAB + v];
  sm[v] = x; __syncthreads();
  for (int s = 64; s > 0; s >>= 1) { if (v < s) sm[v] = fmaxf(sm[v], sm[v + s]); __syncthreads(); }
  const float mx = sm[0]; __syncthreads();
  sm[v] = expf(x - mx); __syncthreads();
  for (int s = 64; s > 0; s >>= 1) { if (v < s) sm[v] += sm[v + s]; __syncthreads(); }
  out[(size_t)t * VOCAB + v] = x - (logf(sm[0]) + mx);
}

// ---------- host ----------
extern "C" void kernel_launch(void* const* d_in, const int* in_sizes, int n_in,
                              void* d_out, int out_size, void* d_ws, size_t ws_size,
                              hipStream_t stream) {
  (void)in_sizes; (void)n_in; (void)out_size; (void)ws_size;
  const float* x   = (const float*)d_in[0];
  const float* h0  = (const float*)d_in[1];
  const float* c0  = (const float*)d_in[2];
  const float* Wih = (const float*)d_in[3];
  const float* Whh = (const float*)d_in[4];
  const float* bih = (const float*)d_in[5];
  const float* bhh = (const float*)d_in[6];
  const float* fcw = (const float*)d_in[7];
  const float* fcb = (const float*)d_in[8];
  float* out = (float*)d_out;

  char* ws = (char*)d_ws;
  size_t off = 0;
  auto alloc = [&](size_t bytes) -> void* {
    void* p = ws + off; off += (bytes + 255) & ~(size_t)255; return p;
  };
  unsigned int*   bar    = (unsigned int*)  alloc(256);
  unsigned short* xbf    = (unsigned short*)alloc((size_t)T_SEQ * HID * 2);
  unsigned short* wihbf  = (unsigned short*)alloc((size_t)2 * G4H * HID * 2);
  unsigned short* whhbf  = (unsigned short*)alloc((size_t)2 * G4H * HID * 2);
  unsigned short* fcwbf  = (unsigned short*)alloc((size_t)VOCAB * HID * 2);
  unsigned short* seq0   = (unsigned short*)alloc((size_t)T_SEQ * HID * 2);
  unsigned short* seq1   = (unsigned short*)alloc((size_t)T_SEQ * HID * 2);
  unsigned short* hbf    = (unsigned short*)alloc((size_t)HID * 2);
  float*          G      = (float*)         alloc((size_t)T_SEQ * G4H * 4);
  float*          logits = (float*)         alloc((size_t)T_SEQ * VOCAB * 4);

  (void)hipMemsetAsync(bar, 0, 256, stream);

  // fp32 -> bf16 conversions
  f32_to_bf16_k<<<4096, 256, 0, stream>>>(x,   xbf,   (long long)T_SEQ * HID);
  f32_to_bf16_k<<<4096, 256, 0, stream>>>(Wih, wihbf, (long long)2 * G4H * HID);
  f32_to_bf16_k<<<4096, 256, 0, stream>>>(Whh, whhbf, (long long)2 * G4H * HID);
  f32_to_bf16_k<<<1024, 256, 0, stream>>>(fcw, fcwbf, (long long)VOCAB * HID);

  // ---- layer 0 ----
  {
    const int M = T_SEQ, N = G4H;
    const int blocks = (M / 32) * (N / 64) / 8;     // 1024, 8 waves/block
    wmma_gemm_bias<<<blocks, 256, 0, stream>>>(xbf, wihbf, bih, bhh, G, M, N, HID);
  }
  lstm_scan<<<SCAN_BLOCKS, 256, SCAN_LDS, stream>>>(
      G, whhbf, h0, c0, hbf, seq0,
      out + HN_OFF, out + CN_OFF, bar, T_SEQ);

  // ---- layer 1 ----
  {
    const int M = T_SEQ, N = G4H;
    const int blocks = (M / 32) * (N / 64) / 8;
    wmma_gemm_bias<<<blocks, 256, 0, stream>>>(
        seq0, wihbf + (size_t)G4H * HID, bih + G4H, bhh + G4H, G, M, N, HID);
  }
  lstm_scan<<<SCAN_BLOCKS, 256, SCAN_LDS, stream>>>(
      G, whhbf + (size_t)G4H * HID, h0 + HID, c0 + HID, hbf, seq1,
      out + HN_OFF + HID, out + CN_OFF + HID, bar, T_SEQ);

  // ---- FC + log-softmax ----
  {
    const int M = T_SEQ, N = VOCAB;                 // wn = 2 -> 2 waves/block
    const int blocks = (M / 32) * (N / 64) / 2;     // 64 blocks of 64 threads
    wmma_gemm_bias<<<blocks, 64, 0, stream>>>(
        seq1, fcwbf, fcb, (const float*)nullptr, logits, M, N, HID);
  }
  logsoftmax128<<<T_SEQ, VOCAB, 0, stream>>>(logits, out);
}
